// SADefnAttn_76768245449528
// MI455X (gfx1250) — matrix operand order repaired and checked by reading, not compiled, not run on hardware
//
#include <hip/hip_runtime.h>
#include <math.h>

// ---------------------------------------------------------------------------
// SADefnAttn for MI455X (gfx1250, wave32, WMMA, async LDS staging).
//   B=2, N=40000, C=128, NH=4, NP=8, DH=32, HS=WS=200.
// Pipeline:
//   prep_q      : q = f16(query + query_pos)                      [80000x128]
//   gemm_wmma x3: value(128) / offsets(64) / attn-logits(32)      (f16 WMMA)
//   gather      : softmax(NP=8) + bilinear sample + head reduce -> f16
//   gemm_wmma   : out1 @ W_out + b_out -> f16 tmp
//   gemm_wmma   : tmp @ W_mlp + b_mlp + residual(query) -> d_out (f32)
// ---------------------------------------------------------------------------

typedef __attribute__((ext_vector_type(16))) _Float16 v16h;
typedef __attribute__((ext_vector_type(8)))  float    v8f;

#define BN      2
#define NQ      40000
#define CC      128
#define NHEAD   4
#define NPOINT  8
#define DHEAD   32
#define HSS     200
#define WSS     200
#define MTOT    (BN * NQ)          // 80000 rows, 80000 = 625 * 128
#define LDSP    136                // padded LDS row stride in halves (272 B)

// gfx1250 async global->LDS DMA (ASYNCcnt-tracked), if the toolchain has it.
#if defined(__has_builtin)
#  if __has_builtin(__builtin_amdgcn_global_load_async_to_lds_b128)
#    define HAVE_ASYNC_LDS 1
#  endif
#endif
#ifndef HAVE_ASYNC_LDS
#  define HAVE_ASYNC_LDS 0
#endif

#if HAVE_ASYNC_LDS
typedef int v4i __attribute__((vector_size(16)));
typedef __attribute__((address_space(1))) v4i* gv4i_p;   // global (AS1)
typedef __attribute__((address_space(3))) v4i* sv4i_p;   // LDS (AS3)
#endif

// ---------------------------------------------------------------------------
// q = f16(query + query_pos), vectorized float4 -> 4 halves
// ---------------------------------------------------------------------------
__global__ __launch_bounds__(256) void prep_q_kernel(
    const float* __restrict__ q, const float* __restrict__ qp,
    _Float16* __restrict__ qh)
{
    int i = blockIdx.x * 256 + threadIdx.x;            // 2,560,000 float4s exactly
    float4 a = ((const float4*)q)[i];
    float4 b = ((const float4*)qp)[i];
    union { _Float16 h[4]; uint2 u; } o;
    o.h[0] = (_Float16)(a.x + b.x);
    o.h[1] = (_Float16)(a.y + b.y);
    o.h[2] = (_Float16)(a.z + b.z);
    o.h[3] = (_Float16)(a.w + b.w);
    ((uint2*)qh)[i] = o.u;
}

// ---------------------------------------------------------------------------
// Generic K=128 WMMA GEMM:  O[M x NOUT] = A_f16[M x 128] * W_f32[128 x NOUT] + b
// Block: 256 threads (8 waves), 128 rows per block. Wave w owns rows
// [w*16, w*16+16) of the tile and sweeps all NOUT/16 column tiles, 4 K-chunks
// of v_wmma_f32_16x16x32_f16 each. Output mode is compile-time: branch-free
// epilogue, 32-bit saddr-form stores. A tile is staged via the gfx1250
// GLOBAL_LOAD_ASYNC_TO_LDS_B128 DMA path (ASYNCcnt) when available.
// ---------------------------------------------------------------------------
template <int NOUT, bool OUTF, bool OUTH, bool RES>
__global__ __launch_bounds__(256) void gemm_wmma_kernel(
    const _Float16* __restrict__ A,
    const float*    __restrict__ W,
    const float*    __restrict__ bias,
    float*          __restrict__ outf,
    _Float16*       __restrict__ outh,
    const float*    __restrict__ resid)
{
    __shared__ __align__(16) _Float16 As[128 * LDSP];   // 34.8 KB
    __shared__ __align__(16) _Float16 Ws[128 * LDSP];   // 34.8 KB (W^T, f16)

    const int tid = threadIdx.x;
    const size_t r0 = (size_t)blockIdx.x * 128;

    // --- stage A tile (f16, row-major, padded): 2048 x 16B, 8 per thread ---
    const uint4* Ag = (const uint4*)(A + r0 * CC);
#if HAVE_ASYNC_LDS
#pragma unroll
    for (int i = 0; i < 8; ++i) {
        int j   = i * 256 + tid;           // 0..2047
        int row = j >> 4, c16 = j & 15;
        __builtin_amdgcn_global_load_async_to_lds_b128(
            (gv4i_p)(Ag + j),
            (sv4i_p)(As + row * LDSP + c16 * 8),
            0, 0);
    }
#else
#pragma unroll
    for (int i = 0; i < 8; ++i) {
        int j   = i * 256 + tid;
        int row = j >> 4, c16 = j & 15;
        *(uint4*)(As + row * LDSP + c16 * 8) = Ag[j];
    }
#endif
    // --- stage W transposed, f32 -> f16 : Ws[n][k] = W[k][n] ---
    constexpr int totW = CC * NOUT;
    for (int idx = tid; idx < totW; idx += 256) {
        int k = idx / NOUT;                // NOUT is a power of two -> shifts
        int n = idx - k * NOUT;
        Ws[n * LDSP + k] = (_Float16)W[idx];
    }
#if HAVE_ASYNC_LDS
    asm volatile("s_wait_asynccnt 0x0" ::: "memory");
#endif
    __syncthreads();

    const int lane = tid & 31, wv = tid >> 5;
    const int hi = lane >> 4, lo = lane & 15;
    const _Float16* arow = As + (wv * 16 + lo) * LDSP;   // A row for this lane

    // Preload all 4 A K-chunk fragments (reused across every column tile).
    union { uint4 u[2]; v16h h; } af[4];
#pragma unroll
    for (int kc = 0; kc < 4; ++kc) {
        // A 16x32 f16 frag: elems 0..7 -> K = kc*32 + hi*8 + e
        //                   elems 8..15 -> K = kc*32 + 16 + hi*8 + (e-8)
        af[kc].u[0] = *(const uint4*)(arow + kc * 32 + hi * 8);
        af[kc].u[1] = *(const uint4*)(arow + kc * 32 + hi * 8 + 16);
    }

    float*       outfB  = OUTF ? outf  + r0 * NOUT : nullptr;
    _Float16*    outhB  = OUTH ? outh  + r0 * NOUT : nullptr;
    const float* residB = RES  ? resid + r0 * NOUT : nullptr;

    constexpr int NTILES = NOUT >> 4;
#pragma unroll
    for (int nt = 0; nt < NTILES; ++nt) {
        v8f acc = {};
        const _Float16* brow = Ws + (nt * 16 + lo) * LDSP;  // B column (W^T row)
#pragma unroll
        for (int kc = 0; kc < 4; ++kc) {
            // B 32x16 f16 frag: elems 0..7 -> K = kc*32 + hi*8 + e, elems 8..15 -> K = kc*32 + 16 + hi*8 + (e-8), N = lo
            union { uint4 u[2]; v16h h; } bf;
            bf.u[0] = *(const uint4*)(brow + kc * 32 + hi * 8);
            bf.u[1] = *(const uint4*)(brow + kc * 32 + hi * 8 + 16);
            acc = __builtin_amdgcn_wmma_f32_16x16x32_f16(
                false, af[kc].h, false, bf.h, (short)0, acc, false, false);
        }
        const int col = nt * 16 + lo;
        const float bv = bias[col];
#pragma unroll
        for (int r = 0; r < 8; ++r) {
            // C/D layout: elem r -> M = r + 8*hi, N = lo
            int li = (wv * 16 + r + hi * 8) * NOUT + col;   // 32-bit tile index
            float v = acc[r] + bv;
            if constexpr (RES)  v += residB[li];
            if constexpr (OUTF) outfB[li] = v;
            if constexpr (OUTH) outhB[li] = (_Float16)v;
        }
    }
}

// ---------------------------------------------------------------------------
// Fused softmax + bilinear gather + head-weighted reduction.
// One wave32 per (b, n, h); lane = head-dim channel. All sampling control
// flow is wave-uniform; each corner read is one coalesced 128 B transaction
// (value tensor is 41 MB -> fully L2 resident on MI455X's 192 MB L2).
// ---------------------------------------------------------------------------
__global__ __launch_bounds__(256) void gather_kernel(
    const float* __restrict__ attn,   // [M x 32]  logits (h*8+p)
    const float* __restrict__ offs,   // [M x 64]  (h*16 + p*2 + c)
    const float* __restrict__ ref,    // [N x 2]
    const float* __restrict__ value,  // [M x 128] (h*32 + d)
    _Float16*    __restrict__ out1)   // [M x 128] f16
{
    int s = blockIdx.x * 8 + (threadIdx.x >> 5);   // (b*N + n)*NH + h
    if (s >= MTOT * NHEAD) return;
    const int lane = threadIdx.x & 31;             // d in [0,32)
    const int h = s & (NHEAD - 1);
    const int n = (s >> 2) % NQ;
    const int b = s / (NQ * NHEAD);
    const size_t m = (size_t)b * NQ + n;

    // softmax over NP=8 (broadcast loads, redundant per lane -- cheap)
    const float* lg = attn + m * (NHEAD * NPOINT) + h * NPOINT;
    float l[NPOINT], mx = -1e30f;
#pragma unroll
    for (int p = 0; p < NPOINT; ++p) { l[p] = lg[p]; mx = fmaxf(mx, l[p]); }
    float sum = 0.f;
#pragma unroll
    for (int p = 0; p < NPOINT; ++p) { l[p] = __expf(l[p] - mx); sum += l[p]; }
    const float inv = 1.0f / sum;

    const float* of = offs + m * (NHEAD * NPOINT * 2) + h * (NPOINT * 2);
    const float rx = ref[2 * n + 0];
    const float ry = ref[2 * n + 1];
    const size_t vbase = (size_t)b * NQ * CC + h * DHEAD + lane;

    float acc = 0.f;
#pragma unroll
    for (int p = 0; p < NPOINT; ++p) {
        float x = (rx + of[2 * p + 0] * (1.0f / WSS)) * WSS - 0.5f;
        float y = (ry + of[2 * p + 1] * (1.0f / HSS)) * HSS - 0.5f;
        float x0f = floorf(x), y0f = floorf(y);
        float dx = x - x0f, dy = y - y0f;
        int x0 = (int)x0f, y0 = (int)y0f;
        const float wp = l[p] * inv;
        float sv = 0.f;
#pragma unroll
        for (int c = 0; c < 4; ++c) {
            int xi = x0 + (c & 1), yi = y0 + (c >> 1);
            float wt = ((c & 1) ? dx : 1.f - dx) * ((c >> 1) ? dy : 1.f - dy);
            if (xi >= 0 && xi < WSS && yi >= 0 && yi < HSS) {
                sv += wt * value[vbase + (size_t)(yi * WSS + xi) * CC];
            }
        }
        acc += wp * sv;
    }
    out1[m * CC + h * DHEAD + lane] = (_Float16)acc;
}

// ---------------------------------------------------------------------------
extern "C" void kernel_launch(void* const* d_in, const int* in_sizes, int n_in,
                              void* d_out, int out_size, void* d_ws, size_t ws_size,
                              hipStream_t stream)
{
    const float* query     = (const float*)d_in[0];
    const float* query_pos = (const float*)d_in[1];
    const float* ref       = (const float*)d_in[2];
    const float* W_off     = (const float*)d_in[3];
    const float* b_off     = (const float*)d_in[4];
    const float* W_attn    = (const float*)d_in[5];
    const float* b_attn    = (const float*)d_in[6];
    const float* W_val     = (const float*)d_in[7];
    const float* b_val     = (const float*)d_in[8];
    const float* W_out     = (const float*)d_in[9];
    const float* b_out     = (const float*)d_in[10];
    const float* W_mlp     = (const float*)d_in[11];
    const float* b_mlp     = (const float*)d_in[12];
    float* out = (float*)d_out;

    // workspace carve-out (all 256B-aligned offsets)
    char* ws = (char*)d_ws;
    _Float16* q_h   = (_Float16*)ws; ws += (size_t)MTOT * CC * 2;   // 20.5 MB
    float*    value = (float*)ws;    ws += (size_t)MTOT * CC * 4;   // 41.0 MB
    float*    offb  = (float*)ws;    ws += (size_t)MTOT * 64 * 4;   // 20.5 MB
    float*    attn  = (float*)ws;    ws += (size_t)MTOT * 32 * 4;   // 10.2 MB
    _Float16* out1h = (_Float16*)ws; ws += (size_t)MTOT * CC * 2;   // 20.5 MB
    _Float16* tmph  = (_Float16*)ws; ws += (size_t)MTOT * CC * 2;   // 20.5 MB

    // 1) q = f16(query + query_pos)
    prep_q_kernel<<<MTOT * CC / 4 / 256, 256, 0, stream>>>(query, query_pos, q_h);

    // 2) three projections off shared q (WMMA f16, f32 accum)
    const int gemmGrid = MTOT / 128;   // 625
    gemm_wmma_kernel<128, true, false, false><<<gemmGrid, 256, 0, stream>>>(
        q_h, W_val, b_val, value, nullptr, nullptr);
    gemm_wmma_kernel<64, true, false, false><<<gemmGrid, 256, 0, stream>>>(
        q_h, W_off, b_off, offb, nullptr, nullptr);
    gemm_wmma_kernel<32, true, false, false><<<gemmGrid, 256, 0, stream>>>(
        q_h, W_attn, b_attn, attn, nullptr, nullptr);

    // 3) softmax + bilinear gather + reduce over points -> f16
    gather_kernel<<<MTOT * NHEAD / 8, 256, 0, stream>>>(attn, offb, ref, value, out1h);

    // 4) output_proj (f16 out, feeds next GEMM), then mlp_out + residual -> f32
    gemm_wmma_kernel<128, false, true, false><<<gemmGrid, 256, 0, stream>>>(
        out1h, W_out, b_out, nullptr, tmph, nullptr);
    gemm_wmma_kernel<128, true, false, true><<<gemmGrid, 256, 0, stream>>>(
        tmph, W_mlp, b_mlp, out, nullptr, query);
}
